// FluxTransformer2DModel_67164698574896
// MI455X (gfx1250) — compile-verified, matrix operand order checked by reading, not run on hardware
//
#include <hip/hip_runtime.h>
#include <math.h>

// ---------------------------------------------------------------------------
// CDNA5 (gfx1250) Flux attention block.
// - All matmuls: V_WMMA_F32_16X16X32_BF16 (f32 accumulate).
// - GEMM tile staging: GLOBAL_LOAD_ASYNC_TO_LDS_B128 (ASYNCcnt fenced).
// - Attention K tile staging: TENSOR_LOAD_TO_LDS (TDM, D# with LDS padding,
//   TENSORcnt fenced). V tile is transposed manually (TDM cannot transpose).
// - Flash attention keeps the 24x2560x2560 score tensor out of HBM.
// - bf16 conversion via native v_cvt; fragments assembled dword-wise.
// ---------------------------------------------------------------------------

typedef __attribute__((ext_vector_type(16))) __bf16        v16bf;
typedef __attribute__((ext_vector_type(8)))  float         v8f;
typedef __attribute__((ext_vector_type(4)))  unsigned int  u32x4;
typedef __attribute__((ext_vector_type(8)))  unsigned int  u32x8;

#define SEQ_ALL 2560
#define SEQ_TXT 512
#define SEQ_IMG 2048
#define DMODEL  3072
#define NHEAD   24
#define DHEAD   128

__device__ __forceinline__ unsigned short f2bf(float x) {
  const __bf16 b = (__bf16)x;          // native v_cvt f32->bf16
  unsigned short u;
  __builtin_memcpy(&u, &b, 2);
  return u;
}

// Pack two floats into one dword of bf16 (lets backend use v_cvt_pk).
__device__ __forceinline__ unsigned int pk2(float a, float b) {
  return (unsigned int)f2bf(a) | ((unsigned int)f2bf(b) << 16);
}

// K-slot pattern per CDNA5 ISA 7.12.2 (16-bit A 16x32): lanes 0-15 take
// K {0..7,16..23}, lanes 16-31 take +8. Each dword i of the operand holds
// bf16 pair (K, K+1).
__device__ __forceinline__ int frag_k(int i, int grp) {
  return 2 * i + ((i < 4) ? 0 : 8) + grp * 8;
}

__device__ __forceinline__ v16bf dwords_to_frag(const u32x8& d) {
  v16bf f;
  __builtin_memcpy(&f, &d, 32);        // free: same VGPRs
  return f;
}

// Fragment from an LDS bf16 tile stored [row][k] (ld ushorts, even).
__device__ __forceinline__ v16bf ld_frag_bf(const unsigned short* base, int ld, int lane) {
  const int grp = (lane >> 4) & 1;
  const int mr  = lane & 15;
  u32x8 d;
#pragma unroll
  for (int i = 0; i < 8; ++i) {
    const int k = frag_k(i, grp);
    unsigned int t;
    __builtin_memcpy(&t, base + mr * ld + k, 4);   // ds_load_b32
    d[i] = t;
  }
  return dwords_to_frag(d);
}

// Fragment from an LDS f32 tile stored [row][k] (ld floats); convert to bf16.
__device__ __forceinline__ v16bf ld_frag_f32row(const float* base, int ld, int lane) {
  const int grp = (lane >> 4) & 1;
  const int mr  = lane & 15;
  u32x8 d;
#pragma unroll
  for (int i = 0; i < 8; ++i) {
    const int k = frag_k(i, grp);
    float p[2];
    __builtin_memcpy(p, base + mr * ld + k, 8);    // ds_load_b64
    d[i] = pk2(p[0], p[1]);
  }
  return dwords_to_frag(d);
}

// Fragment from an LDS f32 tile stored [k][n] (ld floats); n = lane&15.
__device__ __forceinline__ v16bf ld_frag_f32col(const float* base, int ld, int lane) {
  const int grp = (lane >> 4) & 1;
  const int nr  = lane & 15;
  u32x8 d;
#pragma unroll
  for (int i = 0; i < 8; ++i) {
    const int k = frag_k(i, grp);
    d[i] = pk2(base[k * ld + nr], base[(k + 1) * ld + nr]);
  }
  return dwords_to_frag(d);
}

__device__ __forceinline__ v8f vzero8() {
  v8f z = {0.f, 0.f, 0.f, 0.f, 0.f, 0.f, 0.f, 0.f};
  return z;
}

// Per-lane async copy of 16 bytes global -> LDS (no VGPR round trip).
__device__ __forceinline__ void async_b128(unsigned int lds_off, const void* gptr) {
  const unsigned long long ga = (unsigned long long)(uintptr_t)gptr;
  asm volatile("global_load_async_to_lds_b128 %0, %1, off"
               :: "v"(lds_off), "v"(ga) : "memory");
}

__device__ __forceinline__ unsigned int lds_off32(const void* p) {
  // amdgcn generic pointers to LDS carry the AS(3) offset in the low 32 bits.
  return (unsigned int)(uintptr_t)p;
}

// ---------------------------------------------------------------------------
// Generic C[M,N] = A[M,K]*B[K,N] (+bias) with bf16 WMMA.
// Block tile 128x64, BK=32; 8 waves in 4x2 grid, 32x32 per wave (2x2 accums).
// A/B tiles staged as raw f32 via GLOBAL_LOAD_ASYNC_TO_LDS_B128.
// ---------------------------------------------------------------------------
__global__ __launch_bounds__(256) void gemm_bf16(
    const float* __restrict__ A, const float* __restrict__ B,
    const float* __restrict__ bias, float* __restrict__ C,
    int M, int N, int K, int lda, int ldb, int ldc) {
  __shared__ __align__(16) float Af[128 * 36];  // [m][k], pad 4 floats
  __shared__ __align__(16) float Bf[32 * 68];   // [k][n], pad 4 floats

  const int tid  = threadIdx.x;
  const int wave = tid >> 5;
  const int lane = tid & 31;
  const int wm   = wave >> 1;   // 0..3
  const int wn   = wave & 1;    // 0..1
  const int row0 = blockIdx.y * 128;
  const int col0 = blockIdx.x * 64;

  v8f acc[2][2];
  acc[0][0] = vzero8(); acc[0][1] = vzero8();
  acc[1][0] = vzero8(); acc[1][1] = vzero8();

  for (int k0 = 0; k0 < K; k0 += 32) {
    __syncthreads();
    // A: 128x32 f32, 16B per lane per pass, 4 passes
#pragma unroll
    for (int e = 0; e < 4; ++e) {
      const int idx4 = tid + e * 256;                 // 0..1023 float4s
      const int r = idx4 >> 3, c4 = (idx4 & 7) * 4;
      async_b128(lds_off32(&Af[r * 36 + c4]),
                 &A[(size_t)(row0 + r) * lda + k0 + c4]);
    }
    // B: 32x64 f32 row-major [k][n], 2 passes
#pragma unroll
    for (int e = 0; e < 2; ++e) {
      const int idx4 = tid + e * 256;                 // 0..511 float4s
      const int kr = idx4 >> 4, c4 = (idx4 & 15) * 4;
      async_b128(lds_off32(&Bf[kr * 68 + c4]),
                 &B[(size_t)(k0 + kr) * ldb + col0 + c4]);
    }
    if (k0 + 32 < K) {  // global_prefetch_b8 on next tiles
      __builtin_prefetch(&A[(size_t)(row0 + (tid >> 1)) * lda + k0 + 32], 0, 1);
      __builtin_prefetch(&B[(size_t)(k0 + 32 + (tid >> 3)) * ldb + col0 + (tid & 7) * 8], 0, 1);
    }
    asm volatile("s_wait_asynccnt 0x0" ::: "memory");
    __syncthreads();

    const v16bf a0 = ld_frag_f32row(&Af[(wm * 32) * 36],      36, lane);
    const v16bf a1 = ld_frag_f32row(&Af[(wm * 32 + 16) * 36], 36, lane);
    const v16bf b0 = ld_frag_f32col(&Bf[wn * 32],      68, lane);
    const v16bf b1 = ld_frag_f32col(&Bf[wn * 32 + 16], 68, lane);

    acc[0][0] = __builtin_amdgcn_wmma_f32_16x16x32_bf16(false, a0, false, b0, (short)0, acc[0][0], false, false);
    acc[0][1] = __builtin_amdgcn_wmma_f32_16x16x32_bf16(false, a0, false, b1, (short)0, acc[0][1], false, false);
    acc[1][0] = __builtin_amdgcn_wmma_f32_16x16x32_bf16(false, a1, false, b0, (short)0, acc[1][0], false, false);
    acc[1][1] = __builtin_amdgcn_wmma_f32_16x16x32_bf16(false, a1, false, b1, (short)0, acc[1][1], false, false);
  }

  // Store: C/D layout — lane n = lane&15, M = r + 8*(lane>>4)
  const int n = lane & 15, grp = lane >> 4;
#pragma unroll
  for (int sm = 0; sm < 2; ++sm) {
#pragma unroll
    for (int sn = 0; sn < 2; ++sn) {
      const int cc = col0 + wn * 32 + sn * 16 + n;
      const float bv = bias ? bias[cc] : 0.0f;
#pragma unroll
      for (int r = 0; r < 8; ++r) {
        const int rr = row0 + wm * 32 + sm * 16 + grp * 8 + r;
        C[(size_t)rr * ldc + cc] = acc[sm][sn][r] + bv;
      }
    }
  }
}

// ---------------------------------------------------------------------------
// Per-(token, head) RMSNorm + 3-axis RoPE on Q and K in place.
// ---------------------------------------------------------------------------
__global__ __launch_bounds__(128) void rmsrope(
    float* __restrict__ Qb, float* __restrict__ Kb, const int* __restrict__ ids,
    const float* __restrict__ nq_img, const float* __restrict__ nk_img,
    const float* __restrict__ nq_txt, const float* __restrict__ nk_txt) {
  const int t = blockIdx.x, h = blockIdx.y, d = threadIdx.x;
  __shared__ float sq[128], sk[128], rq[128], rk[128];

  float* qp = Qb + (size_t)t * DMODEL + h * DHEAD;
  float* kp = Kb + (size_t)t * DMODEL + h * DHEAD;
  const float q = qp[d];
  const float k = kp[d];
  rq[d] = q * q;
  rk[d] = k * k;
  __syncthreads();
#pragma unroll
  for (int s2 = 64; s2 > 0; s2 >>= 1) {
    if (d < s2) { rq[d] += rq[d + s2]; rk[d] += rk[d + s2]; }
    __syncthreads();
  }
  const float rsq = rsqrtf(rq[0] * (1.0f / 128.0f) + 1e-5f);
  const float rsk = rsqrtf(rk[0] * (1.0f / 128.0f) + 1e-5f);
  const float* qw = (t < SEQ_TXT) ? nq_txt : nq_img;
  const float* kw = (t < SEQ_TXT) ? nk_txt : nk_img;
  sq[d] = q * rsq * qw[d];
  sk[d] = k * rsk * kw[d];
  __syncthreads();

  const int p = d >> 1;
  int axis, j, da;
  if (p < 8)       { axis = 0; j = p;      da = 16; }
  else if (p < 36) { axis = 1; j = p - 8;  da = 56; }
  else             { axis = 2; j = p - 36; da = 56; }
  const float freq = __expf(-9.2103403719761836f * (float)(2 * j) / (float)da);
  const float pos  = (float)ids[t * 3 + axis];
  float s, c;
  __sincosf(pos * freq, &s, &c);

  const float qx1 = sq[2 * p], qx2 = sq[2 * p + 1];
  qp[d] = (d & 1) ? (qx2 * c + qx1 * s) : (qx1 * c - qx2 * s);
  const float kx1 = sk[2 * p], kx2 = sk[2 * p + 1];
  kp[d] = (d & 1) ? (kx2 * c + kx1 * s) : (kx1 * c - kx2 * s);
}

// ---------------------------------------------------------------------------
// Flash attention, one head per blockIdx.y, 128 query rows per block.
// K tile (32x128 f32, LDS row stride 130 dwords via TDM padding) staged with
// TENSOR_LOAD_TO_LDS; V tile transposed manually; P through per-wave LDS.
// ---------------------------------------------------------------------------
__global__ __launch_bounds__(256) void flash_attn(
    const float* __restrict__ Qb, const float* __restrict__ Kb,
    const float* __restrict__ Vb, float* __restrict__ Ob) {
  __shared__ __align__(16) float          Ktf[32 * 130];     // [kv][dh] f32
  __shared__ __align__(16) unsigned short Vt[128 * 34];      // [dh][kv] bf16
  __shared__ __align__(16) unsigned short Pw[8 * 16 * 34];   // per-wave P

  const int tid  = threadIdx.x;
  const int wave = tid >> 5;
  const int lane = tid & 31;
  const int grp  = lane >> 4;
  const int ln   = lane & 15;
  const int h    = blockIdx.y;
  const int t0   = blockIdx.x * 128;
  const float scale = 0.08838834764831845f;  // 1/sqrt(128), folded into Q

  // Preload this wave's 16x128 Q block as 4 bf16 A-fragments.
  const int qrow = t0 + wave * 16 + ln;
  v16bf qa[4];
#pragma unroll
  for (int kk = 0; kk < 4; ++kk) {
    const float* qsrc = Qb + (size_t)qrow * DMODEL + h * DHEAD + kk * 32;
    u32x8 d;
#pragma unroll
    for (int i = 0; i < 8; ++i) {
      const int k = frag_k(i, grp);
      float p[2];
      __builtin_memcpy(p, qsrc + k, 8);  // global_load_b64
      d[i] = pk2(p[0] * scale, p[1] * scale);
    }
    qa[kk] = dwords_to_frag(d);
  }

  v8f o[8];
#pragma unroll
  for (int j = 0; j < 8; ++j) o[j] = vzero8();
  float m_row[8], l_row[8];
#pragma unroll
  for (int r = 0; r < 8; ++r) { m_row[r] = -1e30f; l_row[r] = 0.0f; }

  // TDM descriptor constants for the K tile (uniform).
  const unsigned int kt_lds = lds_off32(&Ktf[0]);

  for (int it = 0; it < SEQ_ALL / 32; ++it) {
    const int kv0 = it * 32;
    __syncthreads();

    // ---- K tile via Tensor Data Mover: 2D tile 128(dh) x 32(kv) f32,
    //      tensor stride 3072 floats, LDS pad 2 dwords per 128 dwords ----
    if (tid == 0) {
      const unsigned long long ga =
          (unsigned long long)(uintptr_t)(Kb + (size_t)kv0 * DMODEL + h * DHEAD);
      u32x4 g0;
      g0[0] = 1u;                                   // count=1
      g0[1] = kt_lds;                               // lds_addr
      g0[2] = (unsigned int)(ga & 0xFFFFFFFFull);   // global_addr[31:0]
      g0[3] = (unsigned int)((ga >> 32) & 0x1FFFFFFull) | (2u << 30);  // +type=2
      u32x8 g1;
      g1[0] = (2u << 16)        // data_size = 4B
            | (1u << 20)        // pad_enable
            | (6u << 22)        // pad_interval: 128 dwords
            | (1u << 25);       // pad_amount: 2 dwords
      g1[1] = (128u << 16);     // tensor_dim0 = 128 (low 16 in bits 63:48)
      g1[2] = (32u << 16);      // tensor_dim0 hi=0 | tensor_dim1 = 32
      g1[3] = (128u << 16);     // tensor_dim1 hi=0 | tile_dim0 = 128
      g1[4] = 32u;              // tile_dim1 = 32, tile_dim2 = 0
      g1[5] = 3072u;            // tensor_dim0_stride lo
      g1[6] = 0u;               // stride hi | tensor_dim1_stride lo
      g1[7] = 0u;
      asm volatile("tensor_load_to_lds %0, %1" :: "s"(g0), "s"(g1) : "memory");
    }

    // ---- V tile: 128-bit loads, transposed bf16 stores ----
#pragma unroll
    for (int e = 0; e < 4; ++e) {
      const int idx4 = tid + e * 256;               // 0..1023 float4s
      const int kv = idx4 >> 5, dh4 = (idx4 & 31) * 4;
      float4 v4;
      __builtin_memcpy(&v4, &Vb[(size_t)(kv0 + kv) * DMODEL + h * DHEAD + dh4], 16);
      Vt[(dh4 + 0) * 34 + kv] = f2bf(v4.x);
      Vt[(dh4 + 1) * 34 + kv] = f2bf(v4.y);
      Vt[(dh4 + 2) * 34 + kv] = f2bf(v4.z);
      Vt[(dh4 + 3) * 34 + kv] = f2bf(v4.w);
    }

    __builtin_amdgcn_s_wait_tensorcnt(0);
    __syncthreads();

    // ---- S = Q K^T : two 16x16 n-tiles over 32 kv cols ----
    v8f s0 = vzero8(), s1 = vzero8();
#pragma unroll
    for (int kk = 0; kk < 4; ++kk) {
      const v16bf b0 = ld_frag_f32row(&Ktf[kk * 32],            130, lane);
      const v16bf b1 = ld_frag_f32row(&Ktf[16 * 130 + kk * 32], 130, lane);
      s0 = __builtin_amdgcn_wmma_f32_16x16x32_bf16(false, qa[kk], false, b0, (short)0, s0, false, false);
      s1 = __builtin_amdgcn_wmma_f32_16x16x32_bf16(false, qa[kk], false, b1, (short)0, s1, false, false);
    }

    // ---- online softmax (row m = r + 8*grp spans the 16 lanes of grp) ----
#pragma unroll
    for (int r = 0; r < 8; ++r) {
      const float v0 = s0[r], v1 = s1[r];
      float mx = fmaxf(v0, v1);
#pragma unroll
      for (int off = 1; off < 16; off <<= 1) mx = fmaxf(mx, __shfl_xor(mx, off, 32));
      const float m_new = fmaxf(m_row[r], mx);
      const float alpha = __expf(m_row[r] - m_new);
      const float p0 = __expf(v0 - m_new);
      const float p1 = __expf(v1 - m_new);
      float rs = p0 + p1;
#pragma unroll
      for (int off = 1; off < 16; off <<= 1) rs += __shfl_xor(rs, off, 32);
      l_row[r] = l_row[r] * alpha + rs;
      m_row[r] = m_new;
#pragma unroll
      for (int j = 0; j < 8; ++j) o[j][r] *= alpha;
      const int m = r + 8 * grp;
      unsigned short* pb = &Pw[wave * 16 * 34 + m * 34];
      pb[ln]      = f2bf(p0);
      pb[16 + ln] = f2bf(p1);
    }
    // same-wave LDS RAW: wait own ds stores before the fragment re-read
    asm volatile("s_wait_dscnt 0" ::: "memory");

    // ---- O += P V : 8 n-tiles of 16 over Dh=128 ----
    const v16bf ap = ld_frag_bf(&Pw[wave * 16 * 34], 34, lane);
#pragma unroll
    for (int j = 0; j < 8; ++j) {
      const v16bf bv = ld_frag_bf(&Vt[j * 16 * 34], 34, lane);
      o[j] = __builtin_amdgcn_wmma_f32_16x16x32_bf16(false, ap, false, bv, (short)0, o[j], false, false);
    }
  }

  // epilogue: O / l
#pragma unroll
  for (int j = 0; j < 8; ++j) {
#pragma unroll
    for (int r = 0; r < 8; ++r) {
      const int m = r + 8 * grp;
      const int token = t0 + wave * 16 + m;
      const int col = h * DHEAD + j * 16 + ln;
      Ob[(size_t)token * DMODEL + col] = o[j][r] / l_row[r];
    }
  }
}

// ---------------------------------------------------------------------------
extern "C" void kernel_launch(void* const* d_in, const int* in_sizes, int n_in,
                              void* d_out, int out_size, void* d_ws, size_t ws_size,
                              hipStream_t stream) {
  (void)in_sizes; (void)n_in; (void)out_size; (void)ws_size;
  const float* hidden     = (const float*)d_in[0];
  const float* enc        = (const float*)d_in[1];
  const int*   ids        = (const int*)d_in[2];
  const float* w_qkv      = (const float*)d_in[3];
  const float* w_add_qkv  = (const float*)d_in[4];
  const float* b_add_qkv  = (const float*)d_in[5];
  const float* w_out      = (const float*)d_in[6];
  const float* b_out      = (const float*)d_in[7];
  const float* w_add_out  = (const float*)d_in[8];
  const float* b_add_out  = (const float*)d_in[9];
  const float* nq         = (const float*)d_in[10];
  const float* nk         = (const float*)d_in[11];
  const float* naq        = (const float*)d_in[12];
  const float* nak        = (const float*)d_in[13];
  float* out = (float*)d_out;
  float* ws  = (float*)d_ws;

  const size_t plane = (size_t)SEQ_ALL * DMODEL;  // 2560*3072 floats
  float* Qb = ws;
  float* Kb = Qb + plane;
  float* Vb = Kb + plane;
  float* Ab = Vb + plane;

  const dim3 blk(256);

  // QKV projections; q/k/v land concatenated [txt(512); img(2048)]
  for (int s = 0; s < 3; ++s) {
    float* dst = (s == 0) ? Qb : (s == 1) ? Kb : Vb;
    gemm_bf16<<<dim3(DMODEL / 64, SEQ_IMG / 128), blk, 0, stream>>>(
        hidden, w_qkv + s * DMODEL, nullptr, dst + (size_t)SEQ_TXT * DMODEL,
        SEQ_IMG, DMODEL, DMODEL, DMODEL, 3 * DMODEL, DMODEL);
    gemm_bf16<<<dim3(DMODEL / 64, SEQ_TXT / 128), blk, 0, stream>>>(
        enc, w_add_qkv + s * DMODEL, b_add_qkv + s * DMODEL, dst,
        SEQ_TXT, DMODEL, DMODEL, DMODEL, 3 * DMODEL, DMODEL);
  }

  rmsrope<<<dim3(SEQ_ALL, NHEAD), dim3(128), 0, stream>>>(Qb, Kb, ids, nq, nk, naq, nak);

  flash_attn<<<dim3(SEQ_ALL / 128, NHEAD), blk, 0, stream>>>(Qb, Kb, Vb, Ab);

  // output projections
  gemm_bf16<<<dim3(DMODEL / 64, SEQ_IMG / 128), blk, 0, stream>>>(
      Ab + (size_t)SEQ_TXT * DMODEL, w_out, b_out, out,
      SEQ_IMG, DMODEL, DMODEL, DMODEL, DMODEL, DMODEL);
  gemm_bf16<<<dim3(DMODEL / 64, SEQ_TXT / 128), blk, 0, stream>>>(
      Ab, w_add_out, b_add_out, out + (size_t)SEQ_IMG * DMODEL,
      SEQ_TXT, DMODEL, DMODEL, DMODEL, DMODEL, DMODEL);
}
